// XEmbedding_72808285602162
// MI455X (gfx1250) — matrix operand order repaired and checked by reading, not compiled, not run on hardware
//
#include <hip/hip_runtime.h>
#include <hip/hip_bf16.h>
#include <math.h>

// ---------------------------------------------------------------------------
// MI455X (gfx1250) implementation.
//  - bf16 WMMA (v_wmma_f32_16x16x32_bf16) for all GEMMs, fp32 accumulate.
//  - Edge kernel is pure streaming (roofline-bound, ~350MB).
//  - Node pipeline: pre-linear (+batchnorm stats) -> stats finalize ->
//    fused norm/MLP/rescale/post-linear/residual.
// ---------------------------------------------------------------------------

typedef __attribute__((ext_vector_type(16))) __bf16 bf16x16;
typedef __attribute__((ext_vector_type(8)))  float  f32x8;

#define NDIM 480
#define NIRR 224

// workspace layout (bytes)
#define WS_SUM     0          // f32[224]  batch-norm sum
#define WS_SUMSQ   896        // f32[224]  batch-norm sum of squares
#define WS_MEAN    1792       // f32[224]
#define WS_INVSTD  2688       // f32[224]
#define WS_W0PRE   4096       // bf16 128x128 (transposed, [v][u])
#define WS_W1PRE   36864      // bf16 64x64
#define WS_W2PRE   45056      // bf16 32x32
#define WS_W0POST  47104      // bf16 128x128
#define WS_W1POST  79872      // bf16 64x64
#define WS_W2POST  88064      // bf16 32x32
#define WS_WMLP    90112      // bf16 224x224
#define W_TOTAL_EL 93184      // total bf16 weight elements

__device__ __forceinline__ f32x8 wmma_bf16(bf16x16 a, bf16x16 b, f32x8 c) {
  // D = A(16x32) * B(32x16) + C(16x16 f32)
  return __builtin_amdgcn_wmma_f32_16x16x32_bf16(false, a, false, b, (short)0, c,
                                                 false, false);
}

// B fragment: B[k,n] = Wt[n][k]; lane L -> n = L&15, k = (j<8?0:16) + 8*(L>>4) + (j&7)
__device__ __forceinline__ bf16x16 bfrag_g(const __bf16* __restrict__ Wt, int ld,
                                           int n, int k0, int half) {
  const __bf16* p = Wt + n * ld + k0 + half * 8;
  bf16x16 b;
#pragma unroll
  for (int j = 0; j < 16; ++j) b[j] = p[((j < 8) ? 0 : 16) + (j & 7)];
  return b;
}

// A fragment gathered from a global fp32 row with element stride `stride`.
// lane L -> m = L&15; element j -> k = k0 + (j<8?0:16) + 8*(L>>4) + (j&7)
__device__ __forceinline__ bf16x16 afrag_gf32(const float* __restrict__ row,
                                              int base, int k0, int stride,
                                              int half) {
  bf16x16 a;
#pragma unroll
  for (int j = 0; j < 16; ++j) {
    int k = k0 + ((j < 8) ? 0 : 16) + half * 8 + (j & 7);
    a[j] = (__bf16)row[base + k * stride];
  }
  return a;
}

// ---------------------------------------------------------------------------
// Weight prep: transpose to [v][u], convert to bf16, zero batchnorm stats.
// ---------------------------------------------------------------------------
__global__ __launch_bounds__(256) void prep_weights(
    const float* __restrict__ W0pre, const float* __restrict__ W1pre,
    const float* __restrict__ W2pre, const float* __restrict__ W0post,
    const float* __restrict__ W1post, const float* __restrict__ W2post,
    const float* __restrict__ Wmlp, char* __restrict__ ws) {
  int t = blockIdx.x * 256 + threadIdx.x;
  if (t < 448) ((float*)ws)[t] = 0.0f;  // zero WS_SUM + WS_SUMSQ
  if (t >= W_TOTAL_EL) return;
  const float* src;
  __bf16* dst;
  int s, e = t;
  if (e < 16384)              { src = W0pre;  dst = (__bf16*)(ws + WS_W0PRE);  s = 128; }
  else if ((e -= 16384) < 4096) { src = W1pre;  dst = (__bf16*)(ws + WS_W1PRE);  s = 64;  }
  else if ((e -= 4096) < 1024)  { src = W2pre;  dst = (__bf16*)(ws + WS_W2PRE);  s = 32;  }
  else if ((e -= 1024) < 16384) { src = W0post; dst = (__bf16*)(ws + WS_W0POST); s = 128; }
  else if ((e -= 16384) < 4096) { src = W1post; dst = (__bf16*)(ws + WS_W1POST); s = 64;  }
  else if ((e -= 4096) < 1024)  { src = W2post; dst = (__bf16*)(ws + WS_W2POST); s = 32;  }
  else { e -= 1024;             src = Wmlp;   dst = (__bf16*)(ws + WS_WMLP);   s = 224; }
  int v = e / s, u = e - v * s;
  dst[e] = (__bf16)src[u * s + v];  // Wt[v][u] = W[u][v]
}

// ---------------------------------------------------------------------------
// Edge kernel: rbf (sin recurrence) + real spherical harmonics l<=2.
// ---------------------------------------------------------------------------
__global__ __launch_bounds__(256) void edge_kernel(
    const float* __restrict__ pos, const int* __restrict__ eidx,
    float* __restrict__ rbf, float* __restrict__ rsh, int E) {
  int e = blockIdx.x * 256 + threadIdx.x;
  if (e >= E) return;
  int si = eidx[e], di = eidx[E + e];
  float ax = pos[3 * si], ay = pos[3 * si + 1], az = pos[3 * si + 2];
  float bx = pos[3 * di], by = pos[3 * di + 1], bz = pos[3 * di + 2];
  // pos2 = pos[:, [1,2,0]]
  float vx = by - ay, vy = bz - az, vz = bx - ax;
  float dist = sqrtf(vx * vx + vy * vy + vz * vz + 1e-24f);
  float invd = 1.0f / dist;
  float dd = dist * 0.1f;  // dist / CUTOFF
  float fc = 0.0f;
  if (dd < 1.0f) {
    float d3 = dd * dd * dd;
    fc = 1.0f + d3 * (-10.0f + dd * (15.0f - 6.0f * dd));
  }
  float pref = 0.44721359549995794f * fc * invd;  // sqrt(2/10)*fc/dist
  float s1, c1;
  __sincosf(3.14159265358979323846f * dd, &s1, &c1);
  float vals[16];
  float sn = s1, cn = c1;
  vals[0] = pref * sn;
#pragma unroll
  for (int n = 1; n < 16; ++n) {  // sin((n+1)x) via angle addition
    float sn2 = sn * c1 + cn * s1;
    float cn2 = cn * c1 - sn * s1;
    sn = sn2; cn = cn2;
    vals[n] = pref * sn;
  }
  float4* rb4 = (float4*)(rbf + (size_t)e * 16);
#pragma unroll
  for (int q = 0; q < 4; ++q)
    rb4[q] = make_float4(vals[4 * q], vals[4 * q + 1], vals[4 * q + 2], vals[4 * q + 3]);
  float ux = -vx * invd, uy = -vy * invd, uz = -vz * invd;
  const float C1 = 0.4886025119029199f, C2 = 1.0925484305920792f;
  float* o = rsh + (size_t)e * 9;
  o[0] = 0.28209479177387814f;
  o[1] = C1 * ux; o[2] = C1 * uy; o[3] = C1 * uz;
  o[4] = C2 * ux * uy;
  o[5] = C2 * uy * uz;
  o[6] = 0.31539156525252005f * (3.0f * uz * uz - 1.0f);
  o[7] = C2 * ux * uz;
  o[8] = 0.5462742152960396f * (ux * ux - uy * uy);
}

// ---------------------------------------------------------------------------
// node_pre: px = irreps_linear(x, W*_pre, b_pre) -> d_out node region,
//           plus per-feature sum/sumsq of invariants (batch-norm stats).
// 8 waves/block, one 16-node tile per wave.
// ---------------------------------------------------------------------------
__global__ __launch_bounds__(256) void node_pre(
    const float* __restrict__ x, const float* __restrict__ bpre,
    float* __restrict__ out, char* __restrict__ ws, int ntiles) {
  int wave = threadIdx.x >> 5;
  int lane = threadIdx.x & 31;
  int half = lane >> 4;
  int l16 = lane & 15;
  int tile = blockIdx.x * 8 + wave;
  if (tile >= ntiles) return;
  int nb = tile * 16;
  const float* row = x + (size_t)(nb + l16) * NDIM;
  float* stat_sum = (float*)(ws + WS_SUM);
  float* stat_sq  = (float*)(ws + WS_SUMSQ);
  const __bf16* Wt0 = (const __bf16*)(ws + WS_W0PRE);
  const __bf16* Wt1 = (const __bf16*)(ws + WS_W1PRE);
  const __bf16* Wt2 = (const __bf16*)(ws + WS_W2PRE);

  // ---- l=0: [16x128] @ [128x128] / sqrt(128) + b ----
  bf16x16 a0[4];
#pragma unroll
  for (int ks = 0; ks < 4; ++ks) a0[ks] = afrag_gf32(row, 0, ks * 32, 1, half);
#pragma unroll
  for (int ch = 0; ch < 8; ++ch) {
    f32x8 acc = {};
#pragma unroll
    for (int ks = 0; ks < 4; ++ks)
      acc = wmma_bf16(a0[ks], bfrag_g(Wt0, 128, ch * 16 + l16, ks * 32, half), acc);
    int v = ch * 16 + l16;
    float bb = bpre[v];
    float s = 0.f, s2 = 0.f;
#pragma unroll
    for (int r = 0; r < 8; ++r) {
      float d = acc[r] * 0.08838834764831845f + bb;  // 1/sqrt(128)
      out[(size_t)(nb + r + half * 8) * NDIM + v] = d;
      float x0 = sqrtf(d * d + 1e-24f);
      s += x0; s2 += x0 * x0;
    }
    s  += __shfl_xor(s, 16, 32);
    s2 += __shfl_xor(s2, 16, 32);
    if (half == 0) { atomicAdd(&stat_sum[v], s); atomicAdd(&stat_sq[v], s2); }
  }

  // ---- l=1: 3x ([16x64] @ [64x64]) / 8 ----
  bf16x16 a1[3][2];
#pragma unroll
  for (int i = 0; i < 3; ++i)
#pragma unroll
    for (int ks = 0; ks < 2; ++ks)
      a1[i][ks] = afrag_gf32(row, 128 + i, ks * 32, 3, half);
#pragma unroll
  for (int ch = 0; ch < 4; ++ch) {
    f32x8 acc0 = {}, acc1 = {}, acc2 = {};
#pragma unroll
    for (int ks = 0; ks < 2; ++ks) {
      bf16x16 bf = bfrag_g(Wt1, 64, ch * 16 + l16, ks * 32, half);
      acc0 = wmma_bf16(a1[0][ks], bf, acc0);
      acc1 = wmma_bf16(a1[1][ks], bf, acc1);
      acc2 = wmma_bf16(a1[2][ks], bf, acc2);
    }
    int v = ch * 16 + l16;
    float s = 0.f, s2 = 0.f;
#pragma unroll
    for (int r = 0; r < 8; ++r) {
      float d0 = acc0[r] * 0.125f, d1 = acc1[r] * 0.125f, d2 = acc2[r] * 0.125f;
      size_t o = (size_t)(nb + r + half * 8) * NDIM + 128 + v * 3;
      out[o] = d0; out[o + 1] = d1; out[o + 2] = d2;
      float x0 = sqrtf(d0 * d0 + d1 * d1 + d2 * d2 + 1e-24f);
      s += x0; s2 += x0 * x0;
    }
    s  += __shfl_xor(s, 16, 32);
    s2 += __shfl_xor(s2, 16, 32);
    if (half == 0) { atomicAdd(&stat_sum[128 + v], s); atomicAdd(&stat_sq[128 + v], s2); }
  }

  // ---- l=2: 5x ([16x32] @ [32x32]) / sqrt(32) ----
  bf16x16 a2[5];
#pragma unroll
  for (int i = 0; i < 5; ++i) a2[i] = afrag_gf32(row, 320 + i, 0, 5, half);
#pragma unroll
  for (int ch = 0; ch < 2; ++ch) {
    f32x8 c0 = {}, c1 = {}, c2 = {}, c3 = {}, c4 = {};
    bf16x16 bf = bfrag_g(Wt2, 32, ch * 16 + l16, 0, half);
    c0 = wmma_bf16(a2[0], bf, c0);
    c1 = wmma_bf16(a2[1], bf, c1);
    c2 = wmma_bf16(a2[2], bf, c2);
    c3 = wmma_bf16(a2[3], bf, c3);
    c4 = wmma_bf16(a2[4], bf, c4);
    int v = ch * 16 + l16;
    float s = 0.f, s2 = 0.f;
#pragma unroll
    for (int r = 0; r < 8; ++r) {
      float d0 = c0[r] * 0.17677669529663687f, d1 = c1[r] * 0.17677669529663687f,
            d2 = c2[r] * 0.17677669529663687f, d3 = c3[r] * 0.17677669529663687f,
            d4 = c4[r] * 0.17677669529663687f;
      size_t o = (size_t)(nb + r + half * 8) * NDIM + 320 + v * 5;
      out[o] = d0; out[o + 1] = d1; out[o + 2] = d2; out[o + 3] = d3; out[o + 4] = d4;
      float x0 = sqrtf(d0 * d0 + d1 * d1 + d2 * d2 + d3 * d3 + d4 * d4 + 1e-24f);
      s += x0; s2 += x0 * x0;
    }
    s  += __shfl_xor(s, 16, 32);
    s2 += __shfl_xor(s2, 16, 32);
    if (half == 0) { atomicAdd(&stat_sum[192 + v], s); atomicAdd(&stat_sq[192 + v], s2); }
  }
}

// ---------------------------------------------------------------------------
__global__ void finalize_stats(char* __restrict__ ws, int nnodes) {
  int t = threadIdx.x;
  if (t < NIRR) {
    float invn = 1.0f / (float)nnodes;
    float mean = ((float*)(ws + WS_SUM))[t] * invn;
    float var  = ((float*)(ws + WS_SUMSQ))[t] * invn - mean * mean;
    ((float*)(ws + WS_MEAN))[t] = mean;
    ((float*)(ws + WS_INVSTD))[t] = rsqrtf(var + 1e-5f);
  }
}

// ---------------------------------------------------------------------------
// node_post: x1 = BN(x0); ns = silu(x1@W_mlp+b); scale = ns/(x0+1e-6);
// node = px + irreps_linear(px*scale, W*_post, b_post).   In place on `px`.
// 4 waves/block, one 16-node tile per wave, per-wave LDS tiles.
// ---------------------------------------------------------------------------
__global__ __launch_bounds__(128) void node_post(
    float* __restrict__ nodeio, const float* __restrict__ gamma,
    const float* __restrict__ beta, const float* __restrict__ bmlp,
    const float* __restrict__ bpost, const char* __restrict__ ws, int ntiles) {
  __shared__ __bf16 x1t[4][16][NIRR];  // x1 tile (A for the MLP GEMM)
  __shared__ __bf16 sct[4][16][NIRR];  // 1/(x0+eps), later *= silu(mlp)
  int wave = threadIdx.x >> 5;
  int lane = threadIdx.x & 31;
  int half = lane >> 4;
  int l16 = lane & 15;
  int tile = blockIdx.x * 4 + wave;
  if (tile >= ntiles) return;
  int nb = tile * 16;
  const float* mean   = (const float*)(ws + WS_MEAN);
  const float* invstd = (const float*)(ws + WS_INVSTD);
  const __bf16* WtM  = (const __bf16*)(ws + WS_WMLP);
  const __bf16* Wt0p = (const __bf16*)(ws + WS_W0POST);
  const __bf16* Wt1p = (const __bf16*)(ws + WS_W1POST);
  const __bf16* Wt2p = (const __bf16*)(ws + WS_W2POST);

  // ---- phase 1: invariants, batch-norm, inverse norms ----
  for (int idx = lane; idx < 16 * NIRR; idx += 32) {
    int m = idx / NIRR, u = idx - m * NIRR;
    const float* pr = nodeio + (size_t)(nb + m) * NDIM;
    float x0;
    if (u < 128) {
      float d = pr[u];
      x0 = sqrtf(d * d + 1e-24f);
    } else if (u < 192) {
      const float* p = pr + 128 + (u - 128) * 3;
      x0 = sqrtf(p[0] * p[0] + p[1] * p[1] + p[2] * p[2] + 1e-24f);
    } else {
      const float* p = pr + 320 + (u - 192) * 5;
      x0 = sqrtf(p[0] * p[0] + p[1] * p[1] + p[2] * p[2] + p[3] * p[3] +
                 p[4] * p[4] + 1e-24f);
    }
    float x1 = gamma[u] * (x0 - mean[u]) * invstd[u] + beta[u];
    x1t[wave][m][u] = (__bf16)x1;
    sct[wave][m][u] = (__bf16)(1.0f / (x0 + 1e-6f));
  }
  asm volatile("s_wait_dscnt 0" ::: "memory");  // same-wave LDS RAW

  // ---- phase 2: new_scalar = silu(x1 @ W_mlp + b_mlp); fold into scale ----
  bf16x16 am[7];
#pragma unroll
  for (int ks = 0; ks < 7; ++ks) {
    bf16x16 a;
#pragma unroll
    for (int j = 0; j < 16; ++j) {
      int k = ks * 32 + ((j < 8) ? 0 : 16) + half * 8 + (j & 7);
      a[j] = x1t[wave][l16][k];
    }
    am[ks] = a;
  }
#pragma unroll
  for (int ch = 0; ch < 14; ++ch) {
    f32x8 acc = {};
#pragma unroll
    for (int ks = 0; ks < 7; ++ks)
      acc = wmma_bf16(am[ks], bfrag_g(WtM, 224, ch * 16 + l16, ks * 32, half), acc);
    int n = ch * 16 + l16;
    float bb = bmlp[n];
#pragma unroll
    for (int r = 0; r < 8; ++r) {
      float z = acc[r] + bb;
      float ns = z / (1.0f + __expf(-z));  // silu
      int m = r + half * 8;
      float iv = (float)sct[wave][m][n];
      sct[wave][m][n] = (__bf16)(iv * ns);
    }
  }
  asm volatile("s_wait_dscnt 0" ::: "memory");

  // ---- phase 3: post irreps-linear on (px * scale), residual, in place ----
  const float* pr = nodeio + (size_t)(nb + l16) * NDIM;
  // l=0 (A frags fully in registers before any store)
  bf16x16 a0[4];
#pragma unroll
  for (int ks = 0; ks < 4; ++ks) {
    bf16x16 a;
#pragma unroll
    for (int j = 0; j < 16; ++j) {
      int k = ks * 32 + ((j < 8) ? 0 : 16) + half * 8 + (j & 7);
      a[j] = (__bf16)(pr[k] * (float)sct[wave][l16][k]);
    }
    a0[ks] = a;
  }
#pragma unroll
  for (int ch = 0; ch < 8; ++ch) {
    f32x8 acc = {};
#pragma unroll
    for (int ks = 0; ks < 4; ++ks)
      acc = wmma_bf16(a0[ks], bfrag_g(Wt0p, 128, ch * 16 + l16, ks * 32, half), acc);
    int v = ch * 16 + l16;
    float bb = bpost[v];
#pragma unroll
    for (int r = 0; r < 8; ++r) {
      size_t o = (size_t)(nb + r + half * 8) * NDIM + v;
      nodeio[o] = nodeio[o] + acc[r] * 0.08838834764831845f + bb;
    }
  }
  // l=1
  bf16x16 a1[3][2];
#pragma unroll
  for (int i = 0; i < 3; ++i)
#pragma unroll
    for (int ks = 0; ks < 2; ++ks) {
      bf16x16 a;
#pragma unroll
      for (int j = 0; j < 16; ++j) {
        int k = ks * 32 + ((j < 8) ? 0 : 16) + half * 8 + (j & 7);
        a[j] = (__bf16)(pr[128 + k * 3 + i] * (float)sct[wave][l16][128 + k]);
      }
      a1[i][ks] = a;
    }
#pragma unroll
  for (int ch = 0; ch < 4; ++ch) {
    f32x8 acc0 = {}, acc1 = {}, acc2 = {};
#pragma unroll
    for (int ks = 0; ks < 2; ++ks) {
      bf16x16 bf = bfrag_g(Wt1p, 64, ch * 16 + l16, ks * 32, half);
      acc0 = wmma_bf16(a1[0][ks], bf, acc0);
      acc1 = wmma_bf16(a1[1][ks], bf, acc1);
      acc2 = wmma_bf16(a1[2][ks], bf, acc2);
    }
    int v = ch * 16 + l16;
#pragma unroll
    for (int r = 0; r < 8; ++r) {
      size_t o = (size_t)(nb + r + half * 8) * NDIM + 128 + v * 3;
      nodeio[o]     += acc0[r] * 0.125f;
      nodeio[o + 1] += acc1[r] * 0.125f;
      nodeio[o + 2] += acc2[r] * 0.125f;
    }
  }
  // l=2
  bf16x16 a2[5];
#pragma unroll
  for (int i = 0; i < 5; ++i) {
    bf16x16 a;
#pragma unroll
    for (int j = 0; j < 16; ++j) {
      int k = ((j < 8) ? 0 : 16) + half * 8 + (j & 7);
      a[j] = (__bf16)(pr[320 + k * 5 + i] * (float)sct[wave][l16][192 + k]);
    }
    a2[i] = a;
  }
#pragma unroll
  for (int ch = 0; ch < 2; ++ch) {
    f32x8 c0 = {}, c1 = {}, c2 = {}, c3 = {}, c4 = {};
    bf16x16 bf = bfrag_g(Wt2p, 32, ch * 16 + l16, 0, half);
    c0 = wmma_bf16(a2[0], bf, c0);
    c1 = wmma_bf16(a2[1], bf, c1);
    c2 = wmma_bf16(a2[2], bf, c2);
    c3 = wmma_bf16(a2[3], bf, c3);
    c4 = wmma_bf16(a2[4], bf, c4);
    int v = ch * 16 + l16;
#pragma unroll
    for (int r = 0; r < 8; ++r) {
      size_t o = (size_t)(nb + r + half * 8) * NDIM + 320 + v * 5;
      nodeio[o]     += c0[r] * 0.17677669529663687f;
      nodeio[o + 1] += c1[r] * 0.17677669529663687f;
      nodeio[o + 2] += c2[r] * 0.17677669529663687f;
      nodeio[o + 3] += c3[r] * 0.17677669529663687f;
      nodeio[o + 4] += c4[r] * 0.17677669529663687f;
    }
  }
}

// ---------------------------------------------------------------------------
extern "C" void kernel_launch(void* const* d_in, const int* in_sizes, int n_in,
                              void* d_out, int out_size, void* d_ws, size_t ws_size,
                              hipStream_t stream) {
  const float* x      = (const float*)d_in[0];
  const float* pos    = (const float*)d_in[1];
  const int*   eidx   = (const int*)d_in[2];
  const float* W0pre  = (const float*)d_in[3];
  const float* W1pre  = (const float*)d_in[4];
  const float* W2pre  = (const float*)d_in[5];
  const float* bpre   = (const float*)d_in[6];
  const float* gamma  = (const float*)d_in[7];
  const float* beta   = (const float*)d_in[8];
  const float* Wmlp   = (const float*)d_in[9];
  const float* bmlp   = (const float*)d_in[10];
  const float* W0post = (const float*)d_in[11];
  const float* W1post = (const float*)d_in[12];
  const float* W2post = (const float*)d_in[13];
  const float* bpost  = (const float*)d_in[14];

  float* out = (float*)d_out;
  char*  ws  = (char*)d_ws;
  int nnodes = in_sizes[0] / NDIM;  // 100000
  int E      = in_sizes[2] / 2;     // 3200000
  float* node = out;
  float* rbf  = out + (size_t)nnodes * NDIM;
  float* rsh  = rbf + (size_t)E * 16;
  int ntiles  = (nnodes + 15) / 16;  // 6250

  prep_weights<<<(W_TOTAL_EL + 255) / 256, 256, 0, stream>>>(
      W0pre, W1pre, W2pre, W0post, W1post, W2post, Wmlp, ws);
  edge_kernel<<<(E + 255) / 256, 256, 0, stream>>>(pos, eidx, rbf, rsh, E);
  node_pre<<<(ntiles + 7) / 8, 256, 0, stream>>>(x, bpre, node, ws, ntiles);
  finalize_stats<<<1, 256, 0, stream>>>(ws, nnodes);
  node_post<<<(ntiles + 3) / 4, 128, 0, stream>>>(node, gamma, beta, bmlp, bpost,
                                                  ws, ntiles);
}
